// AttentionalRNNDecoder_15049565405378
// MI455X (gfx1250) — compile-verified, matrix operand order checked by reading, not run on hardware
//
#include <hip/hip_runtime.h>

// ---------------------------------------------------------------------------
// CDNA5 (gfx1250) attentional GRU decoder.
// All GEMMs run on v_wmma_f32_16x16x32_bf16 (bf16 inputs, f32 accumulate).
// Shapes at every call site are exact multiples of the tiles: no bounds
// checks, 16B vector staging, register + LDS double-buffered pipeline.
// ---------------------------------------------------------------------------

typedef __bf16 bf16;
typedef __attribute__((ext_vector_type(16))) __bf16 v16bf;
typedef __attribute__((ext_vector_type(8)))  float  v8f;

#define BM 64
#define BN 128
#define BK 32
#define LDA_P  (BK + 8)   // 40 halves = 80B rows: 16B-aligned, bank-spread
#define LDB_P  (BK + 8)   // B in [n][k] layout (TRANSB)
#define LDBN_P (BN + 8)   // B in [k][n] layout (non-TRANSB), 136 halves = 272B

// ---------------------------------------------------------------------------
// Tiled WMMA GEMM (shapes must be exact multiples of BM/BN/BK):
//   C[M,N] (+bias[N]) (tanh?) = A[M,K] * op(B)      A,B bf16, accum f32
//   TRANSB=true : B is a weight matrix [N,K] row-major (computes A @ B^T)
//   TRANSB=false: B is [K,N] row-major               (computes A @ B)
//   Optional second A (A2) with kSplit implements A = concat(A, A2) along K
//   (kSplit must be a multiple of BK).
//   Batched via blockIdx.z with element strides (0 => shared operand).
// Block: 256 threads = 8 wave32; wave grid 2(M) x 4(N); wave tile 32x32
// = 2x2 WMMA 16x16x32 fragments. Software pipeline: global->regs for tile
// s+1 overlaps WMMA on tile s; LDS double buffer; one barrier per step.
// ---------------------------------------------------------------------------
template <bool TRANSB, bool TANH, typename CT>
__global__ __launch_bounds__(256) void gemm_wmma(
    const bf16* __restrict__ A, const bf16* __restrict__ A2, int kSplit,
    int ldA, int ldA2,
    const bf16* __restrict__ Bm, int ldB,
    const float* __restrict__ bias,
    CT* __restrict__ C, int ldC,
    int M, int N, int K,
    long long strideA, long long strideB, long long strideC) {
  __shared__ __align__(16) bf16 As[2][BM * LDA_P];
  __shared__ __align__(16) bf16 Bs[2][TRANSB ? (BN * LDB_P) : (BK * LDBN_P)];

  const int tid  = threadIdx.x;
  const int lane = tid & 31;
  const int wave = tid >> 5;
  const int m0   = blockIdx.y * BM;
  const int n0   = blockIdx.x * BN;
  const long long bz = blockIdx.z;

  const bf16* Ab = A + bz * strideA;
  const bf16* Bb = Bm + bz * strideB;
  CT* Cb = C + bz * strideC;

  const int mw = (wave & 1) * 32;   // wave M offset within block tile
  const int nw = (wave >> 1) * 32;  // wave N offset within block tile

  v8f acc[2][2];
#pragma unroll
  for (int i = 0; i < 2; ++i)
#pragma unroll
    for (int j = 0; j < 2; ++j)
#pragma unroll
      for (int e = 0; e < 8; ++e) acc[i][j][e] = 0.0f;

  const int am = lane & 15;            // row (A) / col (B) within 16
  const int kb = (lane & 16) ? 8 : 0;  // K sub-block per documented layout

  // Per-thread staging coordinates.
  const int arow = tid >> 2;           // A: 0..63
  const int aseg = (tid & 3) * 8;      // A: 0,8,16,24 (halves)
  const int bn0  = tid >> 2;           // B (TRANSB): n of segment 0; seg1 = n+64
  const int bs0  = (tid & 3) * 8;
  const int bk0  = tid >> 4;           // B (non-T): k of segment 0; seg1 = k+16
  const int bsn0 = (tid & 15) * 8;

  const bf16* aRow  = Ab + (size_t)(m0 + arow) * ldA + aseg;
  const bf16* aRow2 = (A2 != nullptr)
                          ? (A2 + (size_t)(m0 + arow) * ldA2 + aseg) : nullptr;

  // Global -> registers for K-tile at k0.
  uint4 ra, rb0, rb1;
  auto loadTile = [&](int k0) {
    const int gk = k0 + aseg;          // BK-tiles never straddle kSplit
    const bf16* srcA = (aRow2 != nullptr && gk >= kSplit)
                           ? (aRow2 + (k0 - kSplit)) : (aRow + k0);
    ra = *reinterpret_cast<const uint4*>(srcA);
    if constexpr (TRANSB) {
      rb0 = *reinterpret_cast<const uint4*>(
          Bb + (size_t)(n0 + bn0) * ldB + k0 + bs0);
      rb1 = *reinterpret_cast<const uint4*>(
          Bb + (size_t)(n0 + bn0 + 64) * ldB + k0 + bs0);
    } else {
      rb0 = *reinterpret_cast<const uint4*>(
          Bb + (size_t)(k0 + bk0) * ldB + n0 + bsn0);
      rb1 = *reinterpret_cast<const uint4*>(
          Bb + (size_t)(k0 + bk0 + 16) * ldB + n0 + bsn0);
    }
  };
  // Registers -> LDS buffer.
  auto storeTile = [&](int buf) {
    *reinterpret_cast<uint4*>(&As[buf][arow * LDA_P + aseg]) = ra;
    if constexpr (TRANSB) {
      *reinterpret_cast<uint4*>(&Bs[buf][bn0 * LDB_P + bs0])        = rb0;
      *reinterpret_cast<uint4*>(&Bs[buf][(bn0 + 64) * LDB_P + bs0]) = rb1;
    } else {
      *reinterpret_cast<uint4*>(&Bs[buf][bk0 * LDBN_P + bsn0])        = rb0;
      *reinterpret_cast<uint4*>(&Bs[buf][(bk0 + 16) * LDBN_P + bsn0]) = rb1;
    }
  };

  const int steps = K / BK;
  loadTile(0);
  for (int s = 0; s < steps; ++s) {
    const int buf = s & 1;
    storeTile(buf);
    __syncthreads();
    if (s + 1 < steps) loadTile((s + 1) * BK);      // overlaps WMMA below
    if (A2 == nullptr && s + 2 < steps)             // global_prefetch_b8
      __builtin_prefetch(aRow + (s + 2) * BK, 0, 3);

    // ---- assemble fragments per CDNA5 16-bit WMMA lane layout ----
    v16bf af[2], bfg[2];
#pragma unroll
    for (int v = 0; v < 8; ++v) {
      const int kk = (v < 4) ? (kb + 2 * v) : (16 + kb + 2 * (v - 4));
#pragma unroll
      for (int i = 0; i < 2; ++i) {
        af[i][2 * v]     = As[buf][(mw + i * 16 + am) * LDA_P + kk];
        af[i][2 * v + 1] = As[buf][(mw + i * 16 + am) * LDA_P + kk + 1];
        if constexpr (TRANSB) {
          bfg[i][2 * v]     = Bs[buf][(nw + i * 16 + am) * LDB_P + kk];
          bfg[i][2 * v + 1] = Bs[buf][(nw + i * 16 + am) * LDB_P + kk + 1];
        } else {
          bfg[i][2 * v]     = Bs[buf][kk * LDBN_P + (nw + i * 16 + am)];
          bfg[i][2 * v + 1] = Bs[buf][(kk + 1) * LDBN_P + (nw + i * 16 + am)];
        }
      }
    }

#pragma unroll
    for (int i = 0; i < 2; ++i)
#pragma unroll
      for (int j = 0; j < 2; ++j)
        acc[i][j] = __builtin_amdgcn_wmma_f32_16x16x32_bf16(
            false, af[i], false, bfg[j], (short)0, acc[i][j], false, false);
  }

  // ---- epilogue + store per f32 C/D layout (VGPR r -> rows r / r+8) ----
  const int cn   = lane & 15;
  const int rofs = (lane & 16) ? 8 : 0;
#pragma unroll
  for (int i = 0; i < 2; ++i)
#pragma unroll
    for (int j = 0; j < 2; ++j)
#pragma unroll
      for (int r = 0; r < 8; ++r) {
        const int row = m0 + mw + i * 16 + rofs + r;
        const int col = n0 + nw + j * 16 + cn;
        float vv = acc[i][j][r];
        if (bias) vv += bias[col];
        if (TANH) vv = tanhf(vv);
        Cb[(size_t)row * ldC + col] = (CT)vv;
      }
}

// ---------------------------------------------------------------------------
__global__ void cast_f32_bf16(const float* __restrict__ in,
                              bf16* __restrict__ out, long long n) {
  long long i = (long long)blockIdx.x * blockDim.x + threadIdx.x;
  if (i < n) out[i] = (bf16)in[i];
}

__global__ void zero_h(float* __restrict__ hf, bf16* __restrict__ hb, int n) {
  int i = blockIdx.x * blockDim.x + threadIdx.x;
  if (i < n) { hf[i] = 0.0f; hb[i] = (bf16)0.0f; }
}

// Fused GRU gate update for one time step (f32 gate math).
__global__ void gru_gate_kernel(const bf16* __restrict__ xi,
                                const float* __restrict__ hh,
                                float* __restrict__ h_io,   // read prev, write new
                                bf16* __restrict__ h_bf,
                                bf16* __restrict__ dec_out,
                                int t, int Bn, int Tn, int Hn) {
  int idx = blockIdx.x * blockDim.x + threadIdx.x;
  if (idx >= Bn * Hn) return;
  int b = idx / Hn, j = idx - b * Hn;
  size_t xrow = ((size_t)b * Tn + t) * (3 * Hn);
  float xr = (float)xi[xrow + j];
  float xz = (float)xi[xrow + Hn + j];
  float xn = (float)xi[xrow + 2 * Hn + j];
  size_t hrow = (size_t)b * (3 * Hn);
  float hr = hh[hrow + j];
  float hz = hh[hrow + Hn + j];
  float hn = hh[hrow + 2 * Hn + j];
  float r = 1.0f / (1.0f + __expf(-(xr + hr)));
  float z = 1.0f / (1.0f + __expf(-(xz + hz)));
  float n = tanhf(xn + r * hn);
  float h = (1.0f - z) * n + z * h_io[idx];
  h_io[idx] = h;
  h_bf[idx] = (bf16)h;
  dec_out[((size_t)b * Tn + t) * Hn + j] = (bf16)h;
}

// Row softmax over S with mask, f32 in -> bf16 out. One block per (b,t).
__global__ __launch_bounds__(256) void softmax_kernel(
    const float* __restrict__ scores, const unsigned char* __restrict__ masks,
    bf16* __restrict__ attn, int Tn, int Sn) {
  __shared__ float red[256];
  const int row = blockIdx.x;  // b*T + t
  const int b = row / Tn;
  const float* srow = scores + (size_t)row * Sn;
  const int c0 = threadIdx.x, c1 = threadIdx.x + 256;
  float v0 = -INFINITY, v1 = -INFINITY;
  if (c0 < Sn) v0 = masks[b * Sn + c0] ? -INFINITY : srow[c0];
  if (c1 < Sn) v1 = masks[b * Sn + c1] ? -INFINITY : srow[c1];
  red[threadIdx.x] = fmaxf(v0, v1);
  __syncthreads();
  for (int s = 128; s > 0; s >>= 1) {
    if (threadIdx.x < s) red[threadIdx.x] = fmaxf(red[threadIdx.x], red[threadIdx.x + s]);
    __syncthreads();
  }
  const float rm = red[0];
  __syncthreads();
  float e0 = (c0 < Sn && v0 != -INFINITY) ? __expf(v0 - rm) : 0.0f;
  float e1 = (c1 < Sn && v1 != -INFINITY) ? __expf(v1 - rm) : 0.0f;
  red[threadIdx.x] = e0 + e1;
  __syncthreads();
  for (int s = 128; s > 0; s >>= 1) {
    if (threadIdx.x < s) red[threadIdx.x] += red[threadIdx.x + s];
    __syncthreads();
  }
  const float inv = (red[0] > 0.0f) ? (1.0f / red[0]) : 0.0f;
  if (c0 < Sn) attn[(size_t)row * Sn + c0] = (bf16)(e0 * inv);
  if (c1 < Sn) attn[(size_t)row * Sn + c1] = (bf16)(e1 * inv);
}

__global__ void copy_hlast(const float* __restrict__ h, float* __restrict__ out, int n) {
  int i = blockIdx.x * blockDim.x + threadIdx.x;
  if (i < n) out[i] = h[i];
}

// ---------------------------------------------------------------------------
extern "C" void kernel_launch(void* const* d_in, const int* in_sizes, int n_in,
                              void* d_out, int out_size, void* d_ws, size_t ws_size,
                              hipStream_t stream) {
  (void)in_sizes; (void)n_in; (void)out_size; (void)ws_size;
  constexpr int B = 64, T = 512, S = 512, E = 512, H = 512;
  constexpr int BT = B * T;

  const float* emb   = (const float*)d_in[0];
  const float* enc   = (const float*)d_in[1];
  const unsigned char* masks = (const unsigned char*)d_in[2];
  const float* W_ih  = (const float*)d_in[3];
  const float* W_hh  = (const float*)d_in[4];
  const float* b_ih  = (const float*)d_in[5];
  const float* b_hh  = (const float*)d_in[6];
  const float* W_ao  = (const float*)d_in[7];
  const float* b_ao  = (const float*)d_in[8];
  const float* W_d   = (const float*)d_in[9];
  const float* b_d   = (const float*)d_in[10];
  float* out = (float*)d_out;

  // ---- workspace layout ----
  char* ws = (char*)d_ws;
  size_t off = 0;
  auto take = [&](size_t bytes) -> char* {
    char* p = ws + off;
    off = (off + bytes + 255) & ~(size_t)255;
    return p;
  };
  bf16* emb_bf = (bf16*)take((size_t)BT * E * 2);
  bf16* enc_bf = (bf16*)take((size_t)B * S * H * 2);
  bf16* wih_bf = (bf16*)take((size_t)3 * H * E * 2);
  bf16* whh_bf = (bf16*)take((size_t)3 * H * H * 2);
  bf16* wao_bf = (bf16*)take((size_t)H * 2 * H * 2);
  bf16* wd_bf  = (bf16*)take((size_t)H * H * 2);
  char* xi_region = take((size_t)BT * 3 * H * 2);   // 100 MB, dead after scan
  bf16* xi_bf = (bf16*)xi_region;
  float* hh_f = (float*)take((size_t)B * 3 * H * 4);
  float* h_f  = (float*)take((size_t)B * H * 4);
  bf16*  h_bf = (bf16*)take((size_t)B * H * 2);
  bf16* dec_bf = (bf16*)take((size_t)BT * H * 2);
  float* scores_f = (float*)take((size_t)B * T * S * 4);
  bf16* mix_bf = (bf16*)take((size_t)BT * H * 2);
  // reuse dead xi region after GRU scan:
  bf16* attn_bf = (bf16*)xi_region;                          // 33.5 MB
  bf16* out1_bf = (bf16*)(xi_region + ((size_t)48 << 20));   // past attn

  const dim3 blk(256);
  auto cast = [&](const float* src, bf16* dst, long long n) {
    cast_f32_bf16<<<(unsigned)((n + 255) / 256), 256, 0, stream>>>(src, dst, n);
  };

  // 1) cast inputs/weights to bf16 once
  cast(emb, emb_bf, (long long)BT * E);
  cast(enc, enc_bf, (long long)B * S * H);
  cast(W_ih, wih_bf, 3LL * H * E);
  cast(W_hh, whh_bf, 3LL * H * H);
  cast(W_ao, wao_bf, (long long)H * 2 * H);
  cast(W_d,  wd_bf,  (long long)H * H);
  zero_h<<<(B * H + 255) / 256, 256, 0, stream>>>(h_f, h_bf, B * H);

  // 2) xi = emb @ W_ih^T + b_ih      [BT x 1536]
  {
    dim3 grid((3 * H) / BN, BT / BM, 1);
    gemm_wmma<true, false, bf16><<<grid, blk, 0, stream>>>(
        emb_bf, nullptr, 0, E, 0, wih_bf, E, b_ih, xi_bf, 3 * H,
        BT, 3 * H, E, 0, 0, 0);
  }

  // 3) GRU scan: per step, hh = h @ W_hh^T + b_hh, then fused gates
  for (int t = 0; t < T; ++t) {
    dim3 grid((3 * H) / BN, 1, 1);
    gemm_wmma<true, false, float><<<grid, blk, 0, stream>>>(
        h_bf, nullptr, 0, H, 0, whh_bf, H, b_hh, hh_f, 3 * H,
        B, 3 * H, H, 0, 0, 0);
    gru_gate_kernel<<<(B * H + 255) / 256, 256, 0, stream>>>(
        xi_bf, hh_f, h_f, h_bf, dec_bf, t, B, T, H);
  }

  // 4) scores = dec @ enc^T (batched over B)
  {
    dim3 grid(S / BN, T / BM, B);
    gemm_wmma<true, false, float><<<grid, blk, 0, stream>>>(
        dec_bf, nullptr, 0, H, 0, enc_bf, H, nullptr, scores_f, S,
        T, S, H, (long long)T * H, (long long)S * H, (long long)T * S);
  }

  // 5) masked softmax over S
  softmax_kernel<<<B * T, 256, 0, stream>>>(scores_f, masks, attn_bf, T, S);

  // 6) mix = attn @ enc (batched, B not transposed)
  {
    dim3 grid(H / BN, T / BM, B);
    gemm_wmma<false, false, bf16><<<grid, blk, 0, stream>>>(
        attn_bf, nullptr, 0, S, 0, enc_bf, H, nullptr, mix_bf, H,
        T, H, S, (long long)T * S, (long long)S * H, (long long)T * H);
  }

  // 7) out1 = tanh([mix | dec] @ W_attn_out^T + b)   (virtual concat over K)
  {
    dim3 grid(H / BN, BT / BM, 1);
    gemm_wmma<true, true, bf16><<<grid, blk, 0, stream>>>(
        mix_bf, dec_bf, H, H, H, wao_bf, 2 * H, b_ao, out1_bf, H,
        BT, H, 2 * H, 0, 0, 0);
  }

  // 8) out = out1 @ W_dense^T + b_dense   (f32 output)
  {
    dim3 grid(H / BN, BT / BM, 1);
    gemm_wmma<true, false, float><<<grid, blk, 0, stream>>>(
        out1_bf, nullptr, 0, H, 0, wd_bf, H, b_d, out, H,
        BT, H, H, 0, 0, 0);
  }

  // 9) hidden_states = h_last [1,B,H] appended after out
  copy_hlast<<<(B * H + 255) / 256, 256, 0, stream>>>(h_f, out + (size_t)BT * H, B * H);
}
